// VAEGenerator_48739288875494
// MI455X (gfx1250) — matrix operand order repaired and checked by reading, not compiled, not run on hardware
//
#include <hip/hip_runtime.h>

// ---------------------------------------------------------------------------
// VAE LSTM (B=256, T=512, F=32, H=256, Z=64) for gfx1250 (MI455X, wave32)
// 16 persistent workgroups, one per 16-row batch tile; full enc+dec time loop
// per WG. GEMMs: v_wmma_f32_16x16x32_bf16 (fp32 accumulate). Weights packed
// to bf16 B-fragment streams in d_ws; h state in LDS (bf16); c state in VGPRs.
// Round-4 fix: __builtin_amdgcn_sched_barrier(0) between the 16 B-fragment
// loads and the 8 WMMAs of each k-tile, so the scheduler cannot re-sink the
// loads (which collapsed the pipeline to wait-0 per tile in rounds 3/4).
// ---------------------------------------------------------------------------

typedef __attribute__((ext_vector_type(16))) __bf16 v16bf;
typedef __attribute__((ext_vector_type(8)))  float  v8f;

#if defined(__has_builtin)
#if __has_builtin(__builtin_amdgcn_sched_barrier)
#define SCHED_FENCE() __builtin_amdgcn_sched_barrier(0)
#endif
#endif
#ifndef SCHED_FENCE
#define SCHED_FENCE()
#endif

#define Bsz 256
#define Tsz 512
#define Fsz 32
#define Hsz 256
#define Zsz 64

// ---- packed weight offsets in d_ws (bf16 elements) ----
constexpr size_t OFF_ENC_WIH0 = 0;                                // 1024x32
constexpr size_t OFF_ENC_WHH0 = OFF_ENC_WIH0 + 1024 * 32;         // 1024x256
constexpr size_t OFF_ENC_WIH1 = OFF_ENC_WHH0 + 1024 * 256;        // 1024x256
constexpr size_t OFF_ENC_WHH1 = OFF_ENC_WIH1 + 1024 * 256;        // 1024x256
constexpr size_t OFF_DEC_WIH0 = OFF_ENC_WHH1 + 1024 * 256;        // 1024x32
constexpr size_t OFF_DEC_WHH0 = OFF_DEC_WIH0 + 1024 * 32;         // 1024x256
constexpr size_t OFF_DEC_WIH1 = OFF_DEC_WHH0 + 1024 * 256;        // 1024x256
constexpr size_t OFF_DEC_WHH1 = OFF_DEC_WIH1 + 1024 * 256;        // 1024x256
constexpr size_t OFF_W_MU     = OFF_DEC_WHH1 + 1024 * 256;        // 64x256
constexpr size_t OFF_W_LV     = OFF_W_MU + 64 * 256;              // 64x256
constexpr size_t OFF_W_L2H    = OFF_W_LV + 64 * 256;              // 256x64
constexpr size_t OFF_W_OUT    = OFF_W_L2H + 256 * 64;             // 32x256

// ---------------------------------------------------------------------------
// Prep kernel: pack fp32 weight W[N,K] (row-major, out x in) into bf16
// B-fragment stream. Fragment (ntile,ktile) covers B[k,n] = W[n,k].
// B order used here (16-bit, 32x16; not AMD's table order): lane l holds N = l%16; its 16 contiguous
// bf16 elements are K = 16*(l/16) + e. Packed idx: frag*512 + lane*16 + e.
// ---------------------------------------------------------------------------
__global__ void pack_w_kernel(const float* __restrict__ src,
                              __bf16* __restrict__ dst, int Ndim, int Kdim) {
    int idx = blockIdx.x * 256 + threadIdx.x;
    int total = Ndim * Kdim;
    if (idx >= total) return;
    int frag = idx >> 9;
    int r    = idx & 511;
    int lane = r >> 4;
    int e    = r & 15;
    int nfk  = Kdim >> 5;                 // k-tiles of 32
    int ntile = frag / nfk;
    int ktile = frag - ntile * nfk;
    int n = (lane & 15) + (ntile << 4);
    int k = ((lane >> 4) << 4) + e + (ktile << 5);
    dst[idx] = (__bf16)src[(size_t)n * Kdim + k];
}

// ---------------------------------------------------------------------------
// Helpers
// ---------------------------------------------------------------------------
union FragU { v16bf v; int4 q[2]; };

// Opaque zero: defeats LICM on fragment loads (prevents round-1 VGPR spills)
// without breaking global-addrspace provenance (round-2 flat_load regression).
__device__ __forceinline__ size_t opaque_zero() {
    size_t o = 0;
    asm volatile("" : "+s"(o));
    return o;
}

// A-fragment (16x32 bf16) from [M][K]-major bf16 buffer, row stride ald.
// A-layout: lane l holds M=l%16; e=0..15 -> K=16*(l/16)+e
// (contiguous, matching packed B). Two 16B chunks per lane.
__device__ __forceinline__ v16bf load_afrag(const __bf16* __restrict__ buf,
                                            int ald, int kt, int lane) {
    int m = lane & 15;
    int g = lane >> 4;
    const __bf16* base = buf + (size_t)m * ald + kt * 32 + g * 16;
    FragU u;
    u.q[0] = *(const int4*)(base);
    u.q[1] = *(const int4*)(base + 8);
    return u.v;
}

// B-fragment from packed global stream: 32 contiguous bytes per lane.
__device__ __forceinline__ v16bf load_bfrag(const __bf16* __restrict__ fragbase,
                                            int lane) {
    const __bf16* p = fragbase + lane * 16;
    FragU u;
    u.q[0] = *(const int4*)(p);
    u.q[1] = *(const int4*)(p + 8);
    return u.v;
}

__device__ __forceinline__ v8f wmma_bf16(v16bf a, v16bf b, v8f c) {
    return __builtin_amdgcn_wmma_f32_16x16x32_bf16(
        false, a, false, b, (short)0, c, false, false);
}

__device__ __forceinline__ v8f vbcast(float x) {
    v8f r;
#pragma unroll
    for (int i = 0; i < 8; ++i) r[i] = x;
    return r;
}

__device__ __forceinline__ float sigmf(float x) {
    return 1.f / (1.f + __expf(-x));
}
__device__ __forceinline__ float tanh_f(float x) {
    x = fminf(15.f, fmaxf(-15.f, x));
    float e = __expf(2.f * x);
    return (e - 1.f) / (e + 1.f);
}

// ---------------------------------------------------------------------------
// LSTM gate GEMM: gates[16, 4H] = x[16,32*nktx]*Wih^T + h[16,256]*Whh^T + b.
// Wave w owns N-tiles {16q+2w, 16q+2w+1} for gate block q (i,f,g,o).
// All 8 B-fragments of a k-tile are issued before the 8 WMMAs; a scheduling
// fence pins that order so waits become staggered partial s_wait_loadcnt.
// ---------------------------------------------------------------------------
__device__ __forceinline__ void gate_gemm(const __bf16* __restrict__ Bx, int nktx,
                                          const __bf16* __restrict__ Bh,
                                          const __bf16* __restrict__ xb, int xld,
                                          const __bf16* __restrict__ hb,
                                          const float bias[4][2],
                                          v8f acc[4][2], int w, int lane) {
#pragma unroll
    for (int q = 0; q < 4; ++q)
#pragma unroll
        for (int p = 0; p < 2; ++p) acc[q][p] = vbcast(bias[q][p]);

    // ---- input contribution ----
#pragma unroll 1
    for (int kt = 0; kt < nktx; ++kt) {
        v16bf a = load_afrag(xb, xld, kt, lane);
        const __bf16* base = Bx + (((size_t)(2 * w) * nktx + kt) << 9);
        const size_t qs = ((size_t)16 * nktx) << 9;   // stride per gate block
        const size_t ps = ((size_t)nktx) << 9;        // stride per p
        v16bf bb[8];
#pragma unroll
        for (int i = 0; i < 8; ++i)
            bb[i] = load_bfrag(base + (size_t)(i >> 1) * qs +
                                      (size_t)(i & 1) * ps, lane);
        SCHED_FENCE();                      // keep all 16 loads in flight
#pragma unroll
        for (int i = 0; i < 8; ++i)
            acc[i >> 1][i & 1] = wmma_bf16(a, bb[i], acc[i >> 1][i & 1]);
    }
    // ---- recurrent contribution (K = 256 -> 8 k-tiles) ----
#pragma unroll 1
    for (int kt = 0; kt < 8; ++kt) {
        v16bf a = load_afrag(hb, Hsz, kt, lane);
        const __bf16* base = Bh + (((size_t)(2 * w) * 8 + kt) << 9);
        if (kt < 7) __builtin_prefetch(base + 512, 0, 3);  // next k-slice
        const size_t qs = (size_t)(16 * 8) << 9;
        const size_t ps = (size_t)8 << 9;
        v16bf bb[8];
#pragma unroll
        for (int i = 0; i < 8; ++i)
            bb[i] = load_bfrag(base + (size_t)(i >> 1) * qs +
                                      (size_t)(i & 1) * ps, lane);
        SCHED_FENCE();                      // keep all 16 loads in flight
#pragma unroll
        for (int i = 0; i < 8; ++i)
            acc[i >> 1][i & 1] = wmma_bf16(a, bb[i], acc[i >> 1][i & 1]);
    }
}

// Cell update for this wave's 32 columns; writes new h (bf16) into hb[M][256].
__device__ __forceinline__ void cell_update(const v8f acc[4][2], v8f c[2],
                                            __bf16* __restrict__ hb,
                                            int colbase, int lane) {
#pragma unroll
    for (int p = 0; p < 2; ++p) {
        int col = colbase + 16 * p + (lane & 15);
#pragma unroll
        for (int v = 0; v < 8; ++v) {
            float ig = sigmf(acc[0][p][v]);
            float fg = sigmf(acc[1][p][v]);
            float gg = tanh_f(acc[2][p][v]);
            float og = sigmf(acc[3][p][v]);
            float cn = fg * c[p][v] + ig * gg;
            c[p][v] = cn;
            float h = og * tanh_f(cn);
            int m = v + 8 * (lane >> 4);
            hb[m * Hsz + col] = (__bf16)h;
        }
    }
}

// Single-tile GEMM: D[16,16] = A[16, 32*nkt] * B + bias (B pipelined).
__device__ __forceinline__ v8f gemm_tile(const __bf16* __restrict__ Bp, int nt,
                                         int nkt, const __bf16* __restrict__ abuf,
                                         int ald, float bias, int lane) {
    v8f acc = vbcast(bias);
    v16bf bcur = load_bfrag(Bp + ((size_t)nt * nkt << 9), lane);
#pragma unroll 1
    for (int kt = 0; kt < nkt; ++kt) {
        v16bf a = load_afrag(abuf, ald, kt, lane);
        v16bf bnext = bcur;
        if (kt + 1 < nkt)
            bnext = load_bfrag(Bp + (((size_t)nt * nkt + kt + 1) << 9), lane);
        acc = wmma_bf16(a, bcur, acc);
        bcur = bnext;
    }
    return acc;
}

// ---------------------------------------------------------------------------
// Main persistent kernel: one block = 16 batch rows, 8 waves.
// ---------------------------------------------------------------------------
__global__ __launch_bounds__(256) void vae_lstm_kernel(
    const float* __restrict__ x, const float* __restrict__ eps,
    const float* __restrict__ enc_b0, const float* __restrict__ enc_b1,
    const float* __restrict__ b_mu, const float* __restrict__ b_lv,
    const float* __restrict__ b_l2h, const float* __restrict__ dec_b0,
    const float* __restrict__ dec_b1, const float* __restrict__ b_out,
    const __bf16* __restrict__ wsb, float* __restrict__ out) {
    __shared__ __align__(16) __bf16 h0buf[16 * Hsz];   // layer-0 hidden (bf16)
    __shared__ __align__(16) __bf16 h1buf[16 * Hsz];   // layer-1 hidden (bf16)
    __shared__ __align__(16) __bf16 xbuf[16 * Fsz];    // step input (bf16)
    __shared__ __align__(16) float  mulv[16 * 128];    // mu | logvar (fp32)
    __shared__ __align__(16) __bf16 zbuf[16 * Zsz];    // latent z (bf16)

    const int tid  = threadIdx.x;
    const int lane = tid & 31;
    const int w    = tid >> 5;            // wave 0..7
    const int b0   = blockIdx.x * 16;     // batch-tile base row
    const size_t RECON = (size_t)Bsz * Tsz * Fsz;

    for (int i = tid; i < 16 * Hsz; i += 256) {
        h0buf[i] = (__bf16)0.f;
        h1buf[i] = (__bf16)0.f;
    }

    // per-wave bias registers (column fixed per wave)
    float bias0[4][2], bias1[4][2];
#pragma unroll
    for (int q = 0; q < 4; ++q)
#pragma unroll
        for (int p = 0; p < 2; ++p) {
            int col = (16 * q + 2 * w + p) * 16 + (lane & 15);
            bias0[q][p] = enc_b0[col];
            bias1[q][p] = enc_b1[col];
        }

    v8f c0[2], c1[2];
    c0[0] = vbcast(0.f); c0[1] = vbcast(0.f);
    c1[0] = vbcast(0.f); c1[1] = vbcast(0.f);
    __syncthreads();

    // ===================== encoder: 2-layer LSTM over T ======================
#pragma unroll 1
    for (int t = 0; t < Tsz; ++t) {
        size_t lo = opaque_zero();         // per-iteration opaque offset
        {   // stage x_t tile [16 x 32] -> bf16 LDS
            int m = tid >> 4;
            int col = (tid & 15) << 1;
            const float* xp = x + (size_t)(b0 + m) * (Tsz * Fsz) + (size_t)t * Fsz + col;
            xbuf[m * Fsz + col]     = (__bf16)xp[0];
            xbuf[m * Fsz + col + 1] = (__bf16)xp[1];
        }
        __syncthreads();

        v8f acc[4][2];
        gate_gemm(wsb + OFF_ENC_WIH0 + lo, 1, wsb + OFF_ENC_WHH0 + lo,
                  xbuf, Fsz, h0buf, bias0, acc, w, lane);
        __syncthreads();                       // all reads of old h0 done
        cell_update(acc, c0, h0buf, 32 * w, lane);
        __syncthreads();                       // new h0 visible

        gate_gemm(wsb + OFF_ENC_WIH1 + lo, 8, wsb + OFF_ENC_WHH1 + lo,
                  h0buf, Hsz, h1buf, bias1, acc, w, lane);
        __syncthreads();                       // all reads of old h1 done
        cell_update(acc, c1, h1buf, 32 * w, lane);
        __syncthreads();                       // new h1 visible
    }

    // ===================== latent: mu / logvar / z / hd ======================
    {   // waves 0-3: mu tiles 0-3; waves 4-7: logvar tiles 0-3
        int nt = w & 3;
        const __bf16* Bp = (w < 4) ? (wsb + OFF_W_MU) : (wsb + OFF_W_LV);
        const float* bv  = (w < 4) ? b_mu : b_lv;
        float bias = bv[nt * 16 + (lane & 15)];
        v8f a = gemm_tile(Bp, nt, 8, h1buf, Hsz, bias, lane);
        int N = (lane & 15) + nt * 16;
        size_t gbase = RECON + ((w < 4) ? 0 : (size_t)Bsz * Zsz);
        int ldsoff = (w < 4) ? 0 : 64;
#pragma unroll
        for (int v = 0; v < 8; ++v) {
            int m = v + 8 * (lane >> 4);
            mulv[m * 128 + ldsoff + N] = a[v];
            out[gbase + (size_t)(b0 + m) * Zsz + N] = a[v];
        }
    }
    __syncthreads();
    {   // z = mu + eps * exp(0.5*logvar) -> bf16 [16 x 64]
        int m = tid >> 4;
        int cb = (tid & 15) * 4;
#pragma unroll
        for (int j = 0; j < 4; ++j) {
            int c = cb + j;
            float mu_v = mulv[m * 128 + c];
            float lv_v = mulv[m * 128 + 64 + c];
            float zv = mu_v + eps[(size_t)(b0 + m) * Zsz + c] * __expf(0.5f * lv_v);
            zbuf[m * Zsz + c] = (__bf16)zv;
        }
    }
    __syncthreads();
    {   // hd = z @ W_l2h^T + b; init both decoder hidden layers
#pragma unroll
        for (int p = 0; p < 2; ++p) {
            int nt = 2 * w + p;
            float bias = b_l2h[nt * 16 + (lane & 15)];
            v8f a = gemm_tile(wsb + OFF_W_L2H, nt, 2, zbuf, Zsz, bias, lane);
            int col = nt * 16 + (lane & 15);
#pragma unroll
            for (int v = 0; v < 8; ++v) {
                int m = v + 8 * (lane >> 4);
                __bf16 hb = (__bf16)a[v];
                h0buf[m * Hsz + col] = hb;
                h1buf[m * Hsz + col] = hb;
            }
            c0[p] = vbcast(0.f);
            c1[p] = vbcast(0.f);
        }
    }
    {   // x_in0 = 0
        int i2 = tid * 2;
        xbuf[i2] = (__bf16)0.f;
        xbuf[i2 + 1] = (__bf16)0.f;
    }
#pragma unroll
    for (int q = 0; q < 4; ++q)
#pragma unroll
        for (int p = 0; p < 2; ++p) {
            int col = (16 * q + 2 * w + p) * 16 + (lane & 15);
            bias0[q][p] = dec_b0[col];
            bias1[q][p] = dec_b1[col];
        }
    __syncthreads();

    // ===================== decoder: autoregressive over T ====================
#pragma unroll 1
    for (int t = 0; t < Tsz; ++t) {
        size_t lo = opaque_zero();         // per-iteration opaque offset
        v8f acc[4][2];
        gate_gemm(wsb + OFF_DEC_WIH0 + lo, 1, wsb + OFF_DEC_WHH0 + lo,
                  xbuf, Fsz, h0buf, bias0, acc, w, lane);
        __syncthreads();
        cell_update(acc, c0, h0buf, 32 * w, lane);
        __syncthreads();

        gate_gemm(wsb + OFF_DEC_WIH1 + lo, 8, wsb + OFF_DEC_WHH1 + lo,
                  h0buf, Hsz, h1buf, bias1, acc, w, lane);
        __syncthreads();
        cell_update(acc, c1, h1buf, 32 * w, lane);
        __syncthreads();                       // new h1 visible

        // x_out = h1 @ W_out^T + b_out (F=32 -> 2 tiles, waves 0 & 1)
        if (w < 2) {
            float bias = b_out[w * 16 + (lane & 15)];
            v8f a = gemm_tile(wsb + OFF_W_OUT + lo, w, 8, h1buf, Hsz, bias, lane);
            int N = (lane & 15) + 16 * w;
#pragma unroll
            for (int v = 0; v < 8; ++v) {
                int m = v + 8 * (lane >> 4);
                xbuf[m * Fsz + N] = (__bf16)a[v];                       // feedback
                out[(size_t)(b0 + m) * (Tsz * Fsz) + (size_t)t * Fsz + N] = a[v];
            }
        }
        __syncthreads();                       // x_out visible for next step
    }
}

// ---------------------------------------------------------------------------
// Host launcher
// ---------------------------------------------------------------------------
extern "C" void kernel_launch(void* const* d_in, const int* in_sizes, int n_in,
                              void* d_out, int out_size, void* d_ws, size_t ws_size,
                              hipStream_t stream) {
    (void)in_sizes; (void)n_in; (void)out_size; (void)ws_size;

    const float* x      = (const float*)d_in[0];
    const float* eps    = (const float*)d_in[1];
    const float* eWih0  = (const float*)d_in[2];
    const float* eWhh0  = (const float*)d_in[3];
    const float* eb0    = (const float*)d_in[4];
    const float* eWih1  = (const float*)d_in[5];
    const float* eWhh1  = (const float*)d_in[6];
    const float* eb1    = (const float*)d_in[7];
    const float* Wmu    = (const float*)d_in[8];
    const float* bmu    = (const float*)d_in[9];
    const float* Wlv    = (const float*)d_in[10];
    const float* blv    = (const float*)d_in[11];
    const float* Wl2h   = (const float*)d_in[12];
    const float* bl2h   = (const float*)d_in[13];
    const float* dWih0  = (const float*)d_in[14];
    const float* dWhh0  = (const float*)d_in[15];
    const float* db0    = (const float*)d_in[16];
    const float* dWih1  = (const float*)d_in[17];
    const float* dWhh1  = (const float*)d_in[18];
    const float* db1    = (const float*)d_in[19];
    const float* Wout   = (const float*)d_in[20];
    const float* bout   = (const float*)d_in[21];

    __bf16* wsb = (__bf16*)d_ws;

    struct P { const float* src; size_t off; int N, K; };
    const P tbl[12] = {
        {eWih0, OFF_ENC_WIH0, 1024, 32},  {eWhh0, OFF_ENC_WHH0, 1024, 256},
        {eWih1, OFF_ENC_WIH1, 1024, 256}, {eWhh1, OFF_ENC_WHH1, 1024, 256},
        {dWih0, OFF_DEC_WIH0, 1024, 32},  {dWhh0, OFF_DEC_WHH0, 1024, 256},
        {dWih1, OFF_DEC_WIH1, 1024, 256}, {dWhh1, OFF_DEC_WHH1, 1024, 256},
        {Wmu,   OFF_W_MU,     64,   256}, {Wlv,   OFF_W_LV,     64,   256},
        {Wl2h,  OFF_W_L2H,    256,  64},  {Wout,  OFF_W_OUT,    32,   256},
    };
    for (int i = 0; i < 12; ++i) {
        int total = tbl[i].N * tbl[i].K;
        pack_w_kernel<<<dim3((total + 255) / 256), dim3(256), 0, stream>>>(
            tbl[i].src, wsb + tbl[i].off, tbl[i].N, tbl[i].K);
    }

    vae_lstm_kernel<<<dim3(Bsz / 16), dim3(256), 0, stream>>>(
        x, eps, eb0, eb1, bmu, blv, bl2h, db0, db1, bout, wsb, (float*)d_out);
}
